// SRModel_54236847014054
// MI455X (gfx1250) — matrix operand order repaired and checked by reading, not compiled, not run on hardware
//
#include <hip/hip_runtime.h>
#include <hip/hip_bf16.h>

// ---------------------------------------------------------------------------
// SR model on gfx1250: bf16 WMMA everywhere (encoder convs, fused MLP,
// refinement convs), f32 accumulate via v_wmma_f32_16x16x32_bf16.
// Round 4: fix async global->LDS builtin parameter types (int4 vectors in
// AS1/AS3 per the compiler diagnostic). Zero-init accumulators with bias
// folded into the epilogue; pipelined B-tile preloads; v_tanh/v_exp/v_sin.
// ---------------------------------------------------------------------------

typedef __attribute__((ext_vector_type(16))) __bf16 bf16x16;
typedef __attribute__((ext_vector_type(8)))  float  f32x8;
typedef __attribute__((__vector_size__(16))) int    int4v;

#define H_LR 128
#define W_LR 128
#define H_HR 512
#define W_HR 512
#define NPIX_HR (H_HR * W_HR)

#if __has_builtin(__builtin_amdgcn_global_load_async_to_lds_b128) && \
    __has_builtin(__builtin_amdgcn_s_wait_asynccnt)
#define HAVE_ASYNC_LDS 1
#else
#define HAVE_ASYNC_LDS 0
#endif

enum { MODE_GELU = 0, MODE_RESID = 1, MODE_FINAL = 2 };

__device__ __forceinline__ unsigned short f2bf(float f) {
    __bf16 h = (__bf16)f;
    return __builtin_bit_cast(unsigned short, h);
}
__device__ __forceinline__ float bf2f(unsigned short u) {
    __bf16 h = __builtin_bit_cast(__bf16, u);
    return (float)h;
}
__device__ __forceinline__ float ftanh(float x) {
#if __has_builtin(__builtin_amdgcn_tanhf)
    return __builtin_amdgcn_tanhf(x);   // CDNA5 V_TANH_F32
#else
    float e = __expf(2.0f * x);
    return 1.0f - 2.0f / (e + 1.0f);
#endif
}
__device__ __forceinline__ float gelu_f(float x) {
    float x3 = x * x * x;
    return 0.5f * x * (1.0f + ftanh(0.7978845608028654f * (x + 0.044715f * x3)));
}
__device__ __forceinline__ int clampi(int v, int lo, int hi) {
    return v < lo ? lo : (v > hi ? hi : v);
}
// async global->LDS 16B copy (falls back to VGPR round trip)
__device__ __forceinline__ void copy16_g2l(const unsigned short* gsrc, unsigned short* ldst) {
#if HAVE_ASYNC_LDS
    __builtin_amdgcn_global_load_async_to_lds_b128(
        (__attribute__((address_space(1))) int4v*)gsrc,
        (__attribute__((address_space(3))) int4v*)ldst, 0, 0);
#else
    *(uint4*)ldst = *(const uint4*)gsrc;
#endif
}
__device__ __forceinline__ void stagefill_wait() {
#if HAVE_ASYNC_LDS
    __builtin_amdgcn_s_wait_asynccnt(0);
#endif
}

// ---------------------------------------------------------------------------
// Weight packers: WMMA B-tiles (32x16 bf16, 1KB, lane-major).
// Dense 16-bit B layout: lanes 0-15 -> N=lane, K=0..7,16..23 (2/dword);
// lanes 16-31 -> N=lane-16, K=8..15,24..31.
// ---------------------------------------------------------------------------
__global__ void pack_conv(const float* __restrict__ w, unsigned short* __restrict__ dst,
                          int Cin, int Cout, int nkc, int nob) {
    int idx = blockIdx.x * 256 + threadIdx.x;
    int total = 9 * nkc * nob * 512;
    if (idx >= total) return;
    int e = idx;
    int half = e & 1;  e >>= 1;
    int dw   = e & 7;  e >>= 3;
    int lane = e & 31; e >>= 5;
    int ob = e % nob;  e /= nob;
    int kc = e % nkc;  e /= nkc;
    int off = e;                       // ky*3+kx
    int k   = ((dw < 4) ? 0 : 16) + ((lane < 16) ? 0 : 8) + 2 * (dw & 3) + half;
    int cin = kc * 32 + k;
    int n   = ob * 16 + (lane & 15);
    float v = 0.0f;
    if (cin < Cin && n < Cout) v = w[(size_t)(n * Cin + cin) * 9 + off];
    dst[idx] = f2bf(v);
}

__global__ void pack_mlp(const float* __restrict__ w, unsigned short* __restrict__ dst,
                         int K, int N, int nkc, int nblk) {
    int idx = blockIdx.x * 256 + threadIdx.x;
    int total = nblk * nkc * 512;
    if (idx >= total) return;
    int e = idx;
    int half = e & 1;  e >>= 1;
    int dw   = e & 7;  e >>= 3;
    int lane = e & 31; e >>= 5;
    int kc = e % nkc;  e /= nkc;
    int nb = e;
    int k = kc * 32 + ((dw < 4) ? 0 : 16) + ((lane < 16) ? 0 : 8) + 2 * (dw & 3) + half;
    int n = nb * 16 + (lane & 15);
    float v = (k < K && n < N) ? w[(size_t)k * N + n] : 0.0f;
    dst[idx] = f2bf(v);
}

// ---------------------------------------------------------------------------
// lr (f32 planes) -> [128][128][8] bf16 (ch 3..7 zero)
// ---------------------------------------------------------------------------
__global__ void cvt_lr(const float* __restrict__ lr, unsigned short* __restrict__ lrb) {
    int p = blockIdx.x * 256 + threadIdx.x;
    if (p >= H_LR * W_LR) return;
    unsigned short* o = lrb + (size_t)p * 8;
    o[0] = f2bf(lr[p]);
    o[1] = f2bf(lr[16384 + p]);
    o[2] = f2bf(lr[32768 + p]);
    o[3] = 0; o[4] = 0; o[5] = 0; o[6] = 0; o[7] = 0;
}

// ---------------------------------------------------------------------------
// Refinement input: [512][512][8] bf16 = {bilinear lr_up (3), sr0 (3), 0, 0}
// ---------------------------------------------------------------------------
__global__ void build_x0(const float* __restrict__ lr, const float* __restrict__ sr0f,
                         unsigned short* __restrict__ x0) {
    int p = blockIdx.x * blockDim.x + threadIdx.x;
    if (p >= NPIX_HR) return;
    int y = p >> 9, x = p & 511;
    float sx = (x + 0.5f) * 0.25f - 0.5f;
    float sy = (y + 0.5f) * 0.25f - 0.5f;
    float fx = floorf(sx), fy = floorf(sy);
    float wx = sx - fx, wy = sy - fy;
    int x0i = clampi((int)fx, 0, 127), x1i = clampi((int)fx + 1, 0, 127);
    int y0i = clampi((int)fy, 0, 127), y1i = clampi((int)fy + 1, 0, 127);
    unsigned short* o = x0 + (size_t)p * 8;
#pragma unroll
    for (int c = 0; c < 3; ++c) {
        const float* pl = lr + c * 16384;
        float v00 = pl[y0i * 128 + x0i], v01 = pl[y0i * 128 + x1i];
        float v10 = pl[y1i * 128 + x0i], v11 = pl[y1i * 128 + x1i];
        float v = (1.0f - wy) * ((1.0f - wx) * v00 + wx * v01) +
                  wy * ((1.0f - wx) * v10 + wx * v11);
        o[c]     = f2bf(v);
        o[3 + c] = f2bf(sr0f[(size_t)p * 3 + c]);
    }
    o[6] = 0; o[7] = 0;
}

// ---------------------------------------------------------------------------
// Conv3x3 SAME as 9 shifted GEMMs with WMMA bf16.
// Tile: 16x16 output pixels, 8 waves. LDS stage [18][20][CINP] bf16, filled
// with async global->LDS b128 when available. Each wave computes rows
// {wave, wave+8} for every cout block, reusing B tiles for both rows.
// ---------------------------------------------------------------------------
template <int CINP, int NKC, int NOB>
__global__ __launch_bounds__(256) void conv3x3_wmma(
    const unsigned short* __restrict__ in,      // [H][W][CINP] bf16
    const unsigned short* __restrict__ wpack,   // packed B tiles
    const float* __restrict__ bias, int Cout,
    unsigned short* __restrict__ out, int CoutP,
    const unsigned short* __restrict__ resid,
    const float* __restrict__ sr0f, float* __restrict__ finalout,
    int H, int W, int mode) {
    __shared__ __align__(16) unsigned short stage[18 * 20 * CINP];
    const int tx0 = blockIdx.x * 16, ty0 = blockIdx.y * 16;
    __builtin_prefetch(wpack, 0, 1);
    // ---- stage halo tile (zero pad outside) ----
    for (int pos = threadIdx.x; pos < 18 * 18; pos += 256) {
        int sy = pos / 18, sx = pos - sy * 18;
        int gy = ty0 + sy - 1, gx = tx0 + sx - 1;
        unsigned short* dr = &stage[(sy * 20 + sx) * CINP];
        if ((unsigned)gy < (unsigned)H && (unsigned)gx < (unsigned)W) {
            const unsigned short* sr = &in[((size_t)gy * W + gx) * CINP];
#pragma unroll
            for (int c = 0; c < CINP; c += 8)
                copy16_g2l(sr + c, dr + c);
        } else {
            uint4 z; z.x = z.y = z.z = z.w = 0u;
#pragma unroll
            for (int c = 0; c < CINP; c += 8) *(uint4*)(dr + c) = z;
        }
    }
    stagefill_wait();
    __syncthreads();

    const int wave = threadIdx.x >> 5, lane = threadIdx.x & 31;
    const int mlane = lane & 15;
    const int hisel = (lane < 16) ? 0 : 8;
    const int r0 = wave, r1 = wave + 8;

    auto loadA = [&](int r, int dy, int dx, int kc) -> bf16x16 {
        bf16x16 a;
        const int b0 = kc * 32 + hisel;
        const unsigned short* ap = &stage[((r + dy) * 20 + (mlane + dx)) * CINP + b0];
        uint4 z; z.x = z.y = z.z = z.w = 0u;
        ((uint4*)&a)[0] = (b0 < CINP) ? *(const uint4*)(ap) : z;
        ((uint4*)&a)[1] = (b0 + 16 < CINP) ? *(const uint4*)(ap + 16) : z;
        return a;
    };

    for (int ob = 0; ob < NOB; ++ob) {
        const int n = ob * 16 + mlane;
        const float bv = (n < Cout) ? bias[n] : 0.0f;
        f32x8 acc0 = {}, acc1 = {};     // zero C start; bias folded in epilogue
#pragma unroll
        for (int off = 0; off < 9; ++off) {
            const int dy = off / 3, dx = off - dy * 3;
            // preload this tap's B tiles once, reuse for both rows
            bf16x16 bt[NKC];
#pragma unroll
            for (int kc = 0; kc < NKC; ++kc) {
                const unsigned short* bp =
                    wpack + ((size_t)((off * NKC + kc) * NOB + ob)) * 512 + lane * 16;
                ((uint4*)&bt[kc])[0] = *(const uint4*)(bp);
                ((uint4*)&bt[kc])[1] = *(const uint4*)(bp + 8);
            }
#pragma unroll
            for (int kc = 0; kc < NKC; ++kc)
                acc0 = __builtin_amdgcn_wmma_f32_16x16x32_bf16(
                    false, loadA(r0, dy, dx, kc), false, bt[kc], (short)0, acc0, false, false);
#pragma unroll
            for (int kc = 0; kc < NKC; ++kc)
                acc1 = __builtin_amdgcn_wmma_f32_16x16x32_bf16(
                    false, loadA(r1, dy, dx, kc), false, bt[kc], (short)0, acc1, false, false);
        }
        // ---- epilogue: D lane l -> col n, rows t + (l<16?0:8) ----
        auto epilogue = [&](int r, f32x8& acc) {
            const int y = ty0 + r;
#pragma unroll
            for (int t = 0; t < 8; ++t) {
                const int m = t + hisel;
                const int x = tx0 + m;
                const size_t yx = (size_t)y * W + x;
                float v = acc[t] + bv;
                if (mode == MODE_GELU) v = gelu_f(v);
                else if (mode == MODE_RESID) v += bf2f(resid[yx * CoutP + n]);
                if (mode == MODE_FINAL) {
                    if (n < 3) {
                        float s = v + sr0f[yx * 3 + n];
                        finalout[(size_t)n * ((size_t)H * W) + yx] =
                            fminf(fmaxf(s, 0.0f), 1.0f);
                    }
                } else {
                    out[yx * CoutP + n] = f2bf(v);
                }
            }
        };
        epilogue(r0, acc0);
        epilogue(r1, acc1);
    }
}

// ---------------------------------------------------------------------------
// One MLP output block (16 cols): preload all K-chunk B tiles, then run the
// WMMA chain (zero C start) so loads clause and overlap the matrix pipe.
// ---------------------------------------------------------------------------
template <int NKCC>
__device__ __forceinline__ f32x8 mlp_block(const bf16x16 (&A)[8],
                                           const unsigned short* __restrict__ pk,
                                           int nb, int lane) {
    bf16x16 Bv[NKCC];
#pragma unroll
    for (int c = 0; c < NKCC; ++c) {
        const unsigned short* bp = pk + ((size_t)(nb * NKCC + c)) * 512 + lane * 16;
        ((uint4*)&Bv[c])[0] = *(const uint4*)(bp);
        ((uint4*)&Bv[c])[1] = *(const uint4*)(bp + 8);
    }
    f32x8 acc = {};
#pragma unroll
    for (int c = 0; c < NKCC; ++c)
        acc = __builtin_amdgcn_wmma_f32_16x16x32_bf16(
            false, A[c], false, Bv[c], (short)0, acc, false, false);
    return acc;
}

// ---------------------------------------------------------------------------
// Fused per-pixel MLP + kernel-prediction epilogue.
// 128 threads = 4 waves; each wave owns 16 HR pixels. Hidden state lives in
// registers in WMMA A-layout (8 x bf16x16). D->A relayout per layer via a
// private 16x256 bf16 LDS staging tile per wave.
// ---------------------------------------------------------------------------
struct MlpParams {
    const unsigned short* pack[7];
    const float* bias[7];
};

__global__ __launch_bounds__(128) void mlp_fused(
    const unsigned short* __restrict__ feat,  // [128][128][64] bf16
    const float* __restrict__ lr,             // [3][128][128] f32
    MlpParams P, float* __restrict__ sr0f) {  // [HW][3] f32
    __shared__ __align__(16) unsigned short stg[4 * 16 * 256];
    const int wave = threadIdx.x >> 5, lane = threadIdx.x & 31;
    unsigned short* my = &stg[wave * 4096];
    const int mlane = lane & 15;
    const int hisel = (lane < 16) ? 0 : 8;

    const int p = (blockIdx.x * 4 + wave) * 16 + mlane;
    const int py = p >> 9, px = p & 511;
    const float xlr = px * 0.25f - 0.375f;
    const float ylr = py * 0.25f - 0.375f;
    const float fracx = fminf(fmaxf(xlr - floorf(xlr), 0.0f), 1.0f);
    const float fracy = fminf(fmaxf(ylr - floorf(ylr), 0.0f), 1.0f);
    const int cx = clampi((int)roundf(xlr), 0, 127);
    const int cy = clampi((int)roundf(ylr), 0, 127);
    const float xn = (px + 0.5f) * (1.0f / 256.0f) - 1.0f;
    const float yn = (py + 0.5f) * (1.0f / 256.0f) - 1.0f;
    const unsigned short* featrow = feat + ((size_t)(cy * 128 + cx)) * 64;

    bf16x16 A[8];
    // layer-0 A: chunks 0,1 = contiguous feature b128 loads
#pragma unroll
    for (int c = 0; c < 2; ++c) {
        ((uint4*)&A[c])[0] = *(const uint4*)(featrow + c * 32 + hisel);
        ((uint4*)&A[c])[1] = *(const uint4*)(featrow + c * 32 + hisel + 16);
    }
    // chunks 2,3: positional encoding + frac (element e -> K map per ISA)
#pragma unroll
    for (int c = 2; c < 4; ++c) {
#pragma unroll
        for (int e = 0; e < 16; ++e) {
            int k = ((e < 8) ? 0 : 16) + hisel + (e & 7);
            int f = c * 32 + k;
            float v;
            if (f < 104) {
                int q = f - 64, band = q >> 2, t = q & 3;
                float ang = ((t < 2) ? xn : yn) *
                            (3.14159265358979f * exp2f((float)band));
                v = (t & 1) ? __cosf(ang) : __sinf(ang);
            } else if (f == 104) v = fracx;
            else if (f == 105) v = fracy;
            else v = 0.0f;
            A[c][e] = (__bf16)v;
        }
    }

    auto rebuildA = [&]() {
#pragma unroll
        for (int c = 0; c < 8; ++c) {
            const unsigned short* rp = &my[mlane * 256 + c * 32 + hisel];
            ((uint4*)&A[c])[0] = *(const uint4*)(rp);
            ((uint4*)&A[c])[1] = *(const uint4*)(rp + 16);
        }
    };

    // ---- layer 0 (K=128, 4 chunks) ----
    __builtin_prefetch(P.pack[1], 0, 1);
    {
        const float* bs = P.bias[0];
        for (int nb = 0; nb < 16; ++nb) {
            f32x8 acc = mlp_block<4>(A, P.pack[0], nb, lane);
            float bv = bs[nb * 16 + mlane];
#pragma unroll
            for (int t = 0; t < 8; ++t)
                my[(t + hisel) * 256 + nb * 16 + mlane] = f2bf(gelu_f(acc[t] + bv));
        }
        rebuildA();
    }
    // ---- layers 1..4 (K=256, 8 chunks) ----
    for (int layer = 1; layer < 5; ++layer) {
        __builtin_prefetch(P.pack[layer + 1], 0, 1);
        const float* bs = P.bias[layer];
        for (int nb = 0; nb < 16; ++nb) {
            f32x8 acc = mlp_block<8>(A, P.pack[layer], nb, lane);
            float bv = bs[nb * 16 + mlane];
#pragma unroll
            for (int t = 0; t < 8; ++t)
                my[(t + hisel) * 256 + nb * 16 + mlane] = f2bf(gelu_f(acc[t] + bv));
        }
        rebuildA();
    }
    // ---- kernel logits: 196 cols (13 blocks), no activation ----
    __builtin_prefetch(P.pack[6], 0, 1);
    {
        const float* bs = P.bias[5];
        for (int nb = 0; nb < 13; ++nb) {
            int n = nb * 16 + mlane;
            f32x8 acc = mlp_block<8>(A, P.pack[5], nb, lane);
            float bv = (n < 196) ? bs[n] : 0.0f;
#pragma unroll
            for (int t = 0; t < 8; ++t)
                my[(t + hisel) * 256 + n] = f2bf(acc[t] + bv);
        }
    }
    // ---- gate logits: 4 cols -> staging cols 208..223 ----
    {
        const float* bs = P.bias[6];
        f32x8 acc = mlp_block<8>(A, P.pack[6], 0, lane);
        float bv = (mlane < 4) ? bs[mlane] : 0.0f;
#pragma unroll
        for (int t = 0; t < 8; ++t)
            my[(t + hisel) * 256 + 208 + mlane] = f2bf(acc[t] + bv);
    }

    // ---- per-pixel head epilogue (lanes 0..15, one pixel each) ----
    if (lane < 16) {
        const unsigned short* kl = &my[mlane * 256];
        float g0 = bf2f(kl[208]), g1 = bf2f(kl[209]);
        float g2 = bf2f(kl[210]), g3 = bf2f(kl[211]);
        float gm = fmaxf(fmaxf(g0, g1), fmaxf(g2, g3));
        float e0 = __expf(g0 - gm), e1 = __expf(g1 - gm);
        float e2 = __expf(g2 - gm), e3 = __expf(g3 - gm);
        float gs = e0 + e1 + e2 + e3;
        float r0 = 0.0f, r1 = 0.0f, r2 = 0.0f;
        for (int h = 0; h < 4; ++h) {
            const unsigned short* hk = kl + h * 49;
            float sum = 0.0f, ss = 0.0f;
            for (int k = 0; k < 49; ++k) {
                float v = bf2f(hk[k]);
                sum += v; ss += v * v;
            }
            float mean = sum * (1.0f / 49.0f);
            float var = ss * (1.0f / 49.0f) - mean * mean;
            float inv = 1.0f / (sqrtf(fmaxf(var, 0.0f)) + 1e-6f);
            float zm = -1e30f;
            for (int k = 0; k < 49; ++k)
                zm = fmaxf(zm, (bf2f(hk[k]) - mean) * inv);
            float es = 0.0f;
            for (int k = 0; k < 49; ++k)
                es += __expf((bf2f(hk[k]) - mean) * inv - zm);
            float ge = (h == 0) ? e0 : (h == 1) ? e1 : (h == 2) ? e2 : e3;
            float gh = ge / (gs * es);
            for (int k = 0; k < 49; ++k) {
                float w = __expf((bf2f(hk[k]) - mean) * inv - zm) * gh;
                int kq = k / 7;
                int iy = clampi(cy + kq - 3, 0, 127);
                int ix = clampi(cx + (k - kq * 7) - 3, 0, 127);
                int bi = iy * 128 + ix;
                r0 += w * lr[bi];
                r1 += w * lr[16384 + bi];
                r2 += w * lr[32768 + bi];
            }
        }
        size_t yx = (size_t)py * 512 + px;
        sr0f[yx * 3 + 0] = r0;
        sr0f[yx * 3 + 1] = r1;
        sr0f[yx * 3 + 2] = r2;
    }
}

// ---------------------------------------------------------------------------
// Host orchestration
// ---------------------------------------------------------------------------
extern "C" void kernel_launch(void* const* d_in, const int* in_sizes, int n_in,
                              void* d_out, int out_size, void* d_ws, size_t ws_size,
                              hipStream_t stream) {
    (void)in_sizes; (void)n_in; (void)out_size; (void)ws_size;
    const float* lr = (const float*)d_in[0];
    const float *enc_w[4], *enc_b[4], *mlp_w[7], *mlp_b[7], *ref_w[10], *ref_b[10];
    for (int i = 0; i < 4; ++i) { enc_w[i] = (const float*)d_in[1 + i]; enc_b[i] = (const float*)d_in[5 + i]; }
    for (int i = 0; i < 7; ++i) { mlp_w[i] = (const float*)d_in[9 + i]; mlp_b[i] = (const float*)d_in[16 + i]; }
    for (int i = 0; i < 10; ++i) { ref_w[i] = (const float*)d_in[23 + i]; ref_b[i] = (const float*)d_in[33 + i]; }

    char* base = (char*)d_ws;
    size_t cur = 0;
    auto alloc = [&](size_t bytes) -> char* {
        char* pp = base + cur;
        cur = (cur + bytes + 255) & ~(size_t)255;
        return pp;
    };

    unsigned short* lrb  = (unsigned short*)alloc((size_t)16384 * 8 * 2);
    unsigned short* encA = (unsigned short*)alloc((size_t)16384 * 64 * 2);
    unsigned short* encB = (unsigned short*)alloc((size_t)16384 * 64 * 2);
    float*          sr0f = (float*)alloc((size_t)NPIX_HR * 3 * 4);
    unsigned short* x0   = (unsigned short*)alloc((size_t)NPIX_HR * 8 * 2);
    unsigned short* refA = (unsigned short*)alloc((size_t)NPIX_HR * 48 * 2);
    unsigned short* refB = (unsigned short*)alloc((size_t)NPIX_HR * 48 * 2);

    // packed conv weights: enc0..3 then ref0..9
    int c_cin[14]  = {3, 64, 64, 64, 6, 48, 48, 48, 48, 48, 48, 48, 48, 48};
    int c_cout[14] = {64, 64, 64, 64, 48, 48, 48, 48, 48, 48, 48, 48, 48, 3};
    int c_nkc[14]  = {1, 2, 2, 2, 1, 2, 2, 2, 2, 2, 2, 2, 2, 2};
    int c_nob[14]  = {4, 4, 4, 4, 3, 3, 3, 3, 3, 3, 3, 3, 3, 1};
    const float* c_w[14];
    for (int i = 0; i < 4; ++i) c_w[i] = enc_w[i];
    for (int i = 0; i < 10; ++i) c_w[4 + i] = ref_w[i];
    unsigned short* cpk[14];
    for (int i = 0; i < 14; ++i)
        cpk[i] = (unsigned short*)alloc((size_t)9 * c_nkc[i] * c_nob[i] * 512 * 2);

    int m_nkc[7]  = {4, 8, 8, 8, 8, 8, 8};
    int m_nblk[7] = {16, 16, 16, 16, 16, 13, 1};
    int m_K[7]    = {106, 256, 256, 256, 256, 256, 256};
    int m_N[7]    = {256, 256, 256, 256, 256, 196, 4};
    unsigned short* mpk[7];
    for (int i = 0; i < 7; ++i)
        mpk[i] = (unsigned short*)alloc((size_t)m_nkc[i] * m_nblk[i] * 512 * 2);

    // ---- weight packing + input conversion ----
    cvt_lr<<<64, 256, 0, stream>>>(lr, lrb);
    for (int i = 0; i < 14; ++i) {
        int tot = 9 * c_nkc[i] * c_nob[i] * 512;
        pack_conv<<<(tot + 255) / 256, 256, 0, stream>>>(
            c_w[i], cpk[i], c_cin[i], c_cout[i], c_nkc[i], c_nob[i]);
    }
    for (int i = 0; i < 7; ++i) {
        int tot = m_nkc[i] * m_nblk[i] * 512;
        pack_mlp<<<(tot + 255) / 256, 256, 0, stream>>>(
            mlp_w[i], mpk[i], m_K[i], m_N[i], m_nkc[i], m_nblk[i]);
    }

    // ---- encoder: 4 x conv3x3 + gelu (WMMA) ----
    dim3 gE(8, 8), gR(32, 32);
    conv3x3_wmma<8, 1, 4><<<gE, 256, 0, stream>>>(
        lrb, cpk[0], enc_b[0], 64, encA, 64, nullptr, nullptr, nullptr, 128, 128, MODE_GELU);
    conv3x3_wmma<64, 2, 4><<<gE, 256, 0, stream>>>(
        encA, cpk[1], enc_b[1], 64, encB, 64, nullptr, nullptr, nullptr, 128, 128, MODE_GELU);
    conv3x3_wmma<64, 2, 4><<<gE, 256, 0, stream>>>(
        encB, cpk[2], enc_b[2], 64, encA, 64, nullptr, nullptr, nullptr, 128, 128, MODE_GELU);
    conv3x3_wmma<64, 2, 4><<<gE, 256, 0, stream>>>(
        encA, cpk[3], enc_b[3], 64, encB, 64, nullptr, nullptr, nullptr, 128, 128, MODE_GELU);

    // ---- fused MLP + kernel-prediction -> sr0 ----
    MlpParams P;
    for (int i = 0; i < 7; ++i) { P.pack[i] = mpk[i]; P.bias[i] = mlp_b[i]; }
    mlp_fused<<<NPIX_HR / 64, 128, 0, stream>>>(encB, lr, P, sr0f);

    // ---- refinement CNN (WMMA) ----
    build_x0<<<(NPIX_HR + 255) / 256, 256, 0, stream>>>(lr, sr0f, x0);
    conv3x3_wmma<8, 1, 3><<<gR, 256, 0, stream>>>(
        x0, cpk[4], ref_b[0], 48, refA, 48, nullptr, nullptr, nullptr, 512, 512, MODE_GELU);
    for (int i = 0; i < 4; ++i) {
        conv3x3_wmma<48, 2, 3><<<gR, 256, 0, stream>>>(
            refA, cpk[5 + 2 * i], ref_b[1 + 2 * i], 48, refB, 48,
            nullptr, nullptr, nullptr, 512, 512, MODE_GELU);
        conv3x3_wmma<48, 2, 3><<<gR, 256, 0, stream>>>(
            refB, cpk[6 + 2 * i], ref_b[2 + 2 * i], 48, refA, 48,
            refA, nullptr, nullptr, 512, 512, MODE_RESID);
    }
    conv3x3_wmma<48, 2, 1><<<gR, 256, 0, stream>>>(
        refA, cpk[13], ref_b[9], 3, nullptr, 48,
        nullptr, sr0f, (float*)d_out, 512, 512, MODE_FINAL);
}
